// MultitagAttentionModule_11613591568904
// MI455X (gfx1250) — matrix operand order repaired and checked by reading, not compiled, not run on hardware
//
#include <hip/hip_runtime.h>
#include <hip/hip_bf16.h>
#include <cmath>

#define N_TOTAL 32768
#define DIM     2048
#define DPRIME  512
#define NCLASSES 4
#define NCHUNKS 64   // n-chunks for pooling partials

typedef __attribute__((ext_vector_type(16))) __bf16 v16bf;
typedef __attribute__((ext_vector_type(8)))  float  v8f;

// Hardware transcendental tanh (CDNA5 V_TANH_F32). Trailing v_nop covers the
// TRANS-op result hazard (ISA 7.4: 1 independent op / V_NOP before use).
__device__ __forceinline__ float fast_tanh(float x) {
  float r;
  asm("v_tanh_f32 %0, %1\n\tv_nop" : "=v"(r) : "v"(x));
  return r;
}

// ---- WMMA fragment loaders ----------------------------------------------
// A-matrix 16x32 bf16 layout (ISA 7.12.2): lane l holds row l%16,
// K = { 8*(l/16) .. +7 } in v[0..7] halves and { 16 + 8*(l/16) .. +7 } in v[8..15].
__device__ __forceinline__ v16bf load_a_frag_f32(const float* __restrict__ base) {
  // base already includes row*DIM + k + 8*(lane>=16); segments at +0 and +16
  const float4* p0 = reinterpret_cast<const float4*>(base);
  const float4* p1 = reinterpret_cast<const float4*>(base + 16);
  float4 a0 = p0[0], a1 = p0[1];
  float4 b0 = p1[0], b1 = p1[1];
  v16bf r;
  r[0]=(__bf16)a0.x; r[1]=(__bf16)a0.y; r[2]=(__bf16)a0.z; r[3]=(__bf16)a0.w;
  r[4]=(__bf16)a1.x; r[5]=(__bf16)a1.y; r[6]=(__bf16)a1.z; r[7]=(__bf16)a1.w;
  r[8]=(__bf16)b0.x; r[9]=(__bf16)b0.y; r[10]=(__bf16)b0.z; r[11]=(__bf16)b0.w;
  r[12]=(__bf16)b1.x; r[13]=(__bf16)b1.y; r[14]=(__bf16)b1.z; r[15]=(__bf16)b1.w;
  return r;
}

// B-matrix 32x16 bf16 layout: lane l holds column l%16, K = 8*(l/16) + {0..7}
// in v[0..7] and 16 + 8*(l/16) + {0..7} in v[8..15]. B[k][n] = V'[n][k].
__device__ __forceinline__ v16bf load_b_frag(const float* __restrict__ base) {
  const float4* p = reinterpret_cast<const float4*>(base);
  float4 a0 = p[0], a1 = p[1], a2 = p[4], a3 = p[5];
  v16bf r;
  r[0]=(__bf16)a0.x; r[1]=(__bf16)a0.y; r[2]=(__bf16)a0.z; r[3]=(__bf16)a0.w;
  r[4]=(__bf16)a1.x; r[5]=(__bf16)a1.y; r[6]=(__bf16)a1.z; r[7]=(__bf16)a1.w;
  r[8]=(__bf16)a2.x; r[9]=(__bf16)a2.y; r[10]=(__bf16)a2.z; r[11]=(__bf16)a2.w;
  r[12]=(__bf16)a3.x; r[13]=(__bf16)a3.y; r[14]=(__bf16)a3.z; r[15]=(__bf16)a3.w;
  return r;
}

// ---- K0: pack U_t (f32, row-major) into bf16 WMMA-A fragment order -------
// Abuf[((mt_g*64 + ks)*32 + lane)*16 + i] = A-fragment element i of lane
// `lane` for M-tile mt_g at K-step ks. One thread per (mt_g, ks, lane).
__global__ void __launch_bounds__(256) k_pack_ut(
    const float* __restrict__ Ut, unsigned short* __restrict__ AbufU16)
{
  const int idx  = blockIdx.x * 256 + threadIdx.x;  // 0 .. 32*64*32-1
  const int lane = idx & 31;
  const int ks   = (idx >> 5) & 63;
  const int mt_g = idx >> 11;
  const int m  = mt_g * 16 + (lane & 15);
  const int kh = (lane >> 4) * 8;
  const float* src = Ut + (size_t)m * DIM + ks * 32 + kh;
  __bf16* dst = reinterpret_cast<__bf16*>(AbufU16) + (size_t)idx * 16;
  #pragma unroll
  for (int i = 0; i < 8; ++i) dst[i]     = (__bf16)src[i];
  #pragma unroll
  for (int i = 0; i < 8; ++i) dst[8 + i] = (__bf16)src[16 + i];
}

// ---- K1: fused  z[n] = sum_m W_mta[m] * tanh( (U_t @ V'^T)[m,n] ) --------
// Grid: N/64 blocks of 256 threads (8 waves). Wave w owns M rows [64w,64w+64),
// all 64 columns of the block: 4x4 WMMA accumulator tiles (128 VGPRs).
template <bool USE_ABUF>
__global__ void __launch_bounds__(256) k_gemm_tanh_z(
    const float* __restrict__ Vp,            // [N, D]
    const float* __restrict__ Ut,            // [D', D]  (fallback path)
    const unsigned short* __restrict__ Abuf, // packed bf16 fragments
    const float* __restrict__ Wmta,          // [D']
    float* __restrict__ z)                   // [N]
{
  const int lane  = threadIdx.x & 31;
  const int wave  = threadIdx.x >> 5;
  const int lo16  = lane & 15;
  const int hi    = lane >> 4;          // 0 or 1
  const int nbase = blockIdx.x * 64;
  const int mbase = wave * 64;

  v8f acc[4][4];
  #pragma unroll
  for (int mt = 0; mt < 4; ++mt)
    #pragma unroll
    for (int nt = 0; nt < 4; ++nt)
      acc[mt][nt] = (v8f){0.f,0.f,0.f,0.f,0.f,0.f,0.f,0.f};

  const float* bptr[4];
  #pragma unroll
  for (int nt = 0; nt < 4; ++nt)
    bptr[nt] = Vp + (size_t)(nbase + nt*16 + lo16) * DIM + hi*8;
  const float* aptr[4];
  #pragma unroll
  for (int mt = 0; mt < 4; ++mt)
    aptr[mt] = Ut + (size_t)(mbase + mt*16 + lo16) * DIM + hi*8;
  const v16bf* afrag_ptr[4];
  #pragma unroll
  for (int mt = 0; mt < 4; ++mt)
    afrag_ptr[mt] = reinterpret_cast<const v16bf*>(
        Abuf + ((size_t)(wave*4 + mt) * 64 * 32 + lane) * 16);

  for (int ks = 0; ks < DIM/32; ++ks) {
    v16bf bf[4];
    #pragma unroll
    for (int nt = 0; nt < 4; ++nt) bf[nt] = load_b_frag(bptr[nt] + ks*32);
    #pragma unroll
    for (int mt = 0; mt < 4; ++mt) {
      v16bf af;
      if constexpr (USE_ABUF) af = afrag_ptr[mt][ks * 32];    // 32B/lane direct
      else                    af = load_a_frag_f32(aptr[mt] + ks*32);
      #pragma unroll
      for (int nt = 0; nt < 4; ++nt)
        acc[mt][nt] = __builtin_amdgcn_wmma_f32_16x16x32_bf16(
            false, af, false, bf[nt], (short)0, acc[mt][nt], false, false);
    }
  }

  // C/D layout: acc[mt][nt][r] = S[mbase + mt*16 + 8*hi + r, nbase + nt*16 + lo16]
  float wv[4][8];
  #pragma unroll
  for (int mt = 0; mt < 4; ++mt) {
    const float4* wp = reinterpret_cast<const float4*>(Wmta + mbase + mt*16 + hi*8);
    float4 w0 = wp[0], w1 = wp[1];
    wv[mt][0]=w0.x; wv[mt][1]=w0.y; wv[mt][2]=w0.z; wv[mt][3]=w0.w;
    wv[mt][4]=w1.x; wv[mt][5]=w1.y; wv[mt][6]=w1.z; wv[mt][7]=w1.w;
  }

  __shared__ float zred[8][64];
  #pragma unroll
  for (int nt = 0; nt < 4; ++nt) {
    float zt = 0.f;
    #pragma unroll
    for (int mt = 0; mt < 4; ++mt)
      #pragma unroll
      for (int r = 0; r < 8; ++r)
        zt = fmaf(wv[mt][r], fast_tanh(acc[mt][nt][r]), zt);
    zt += __shfl_xor(zt, 16, 32);       // combine M and M+8 halves (same n)
    if (hi == 0) zred[wave][nt*16 + lo16] = zt;
  }
  __syncthreads();
  if (threadIdx.x < 64) {
    float s = 0.f;
    #pragma unroll
    for (int w = 0; w < 8; ++w) s += zred[w][threadIdx.x];
    z[nbase + threadIdx.x] = s;
  }
}

// ---- K2: softmax over N (single block; N=32768) --------------------------
__global__ void __launch_bounds__(1024) k_softmax(
    const float* __restrict__ z, float* __restrict__ a)
{
  __shared__ float red[1024];
  const int tid = threadIdx.x;
  float m = -INFINITY;
  for (int i = tid; i < N_TOTAL; i += 1024) m = fmaxf(m, z[i]);
  red[tid] = m; __syncthreads();
  for (int s = 512; s > 0; s >>= 1) {
    if (tid < s) red[tid] = fmaxf(red[tid], red[tid + s]);
    __syncthreads();
  }
  const float zmax = red[0];
  __syncthreads();
  float sum = 0.f;
  for (int i = tid; i < N_TOTAL; i += 1024) sum += __expf(z[i] - zmax);
  red[tid] = sum; __syncthreads();
  for (int s = 512; s > 0; s >>= 1) {
    if (tid < s) red[tid] += red[tid + s];
    __syncthreads();
  }
  const float inv = 1.0f / red[0];
  for (int i = tid; i < N_TOTAL; i += 1024) a[i] = __expf(z[i] - zmax) * inv;
}

// ---- K3: t_partial[chunk][d] = sum_{n in chunk} a[n] * V'[n,d] -----------
__global__ void __launch_bounds__(256) k_pool(
    const float* __restrict__ Vp, const float* __restrict__ a,
    float* __restrict__ tpart)
{
  const int d  = blockIdx.x * 256 + threadIdx.x;     // coalesced over D
  const int n0 = blockIdx.y * (N_TOTAL / NCHUNKS);
  float acc = 0.f;
  for (int i = 0; i < N_TOTAL / NCHUNKS; ++i) {
    const int n = n0 + i;
    acc = fmaf(a[n], Vp[(size_t)n * DIM + d], acc);
  }
  tpart[(size_t)blockIdx.y * DIM + d] = acc;
}

// ---- K4: reduce partials to t[D], then out[c] = sum_d W[c,d]*t[d] --------
__global__ void __launch_bounds__(256) k_final(
    const float* __restrict__ tpart, const float* __restrict__ W,
    float* __restrict__ out)
{
  const int tid = threadIdx.x;
  float pc[NCLASSES] = {0.f, 0.f, 0.f, 0.f};
  for (int d = tid; d < DIM; d += 256) {
    float tv = 0.f;
    for (int j = 0; j < NCHUNKS; ++j) tv += tpart[(size_t)j * DIM + d];
    #pragma unroll
    for (int c = 0; c < NCLASSES; ++c)
      pc[c] = fmaf(W[(size_t)c * DIM + d], tv, pc[c]);
  }
  __shared__ float red[NCLASSES][256];
  #pragma unroll
  for (int c = 0; c < NCLASSES; ++c) red[c][tid] = pc[c];
  __syncthreads();
  for (int s = 128; s > 0; s >>= 1) {
    if (tid < s) {
      #pragma unroll
      for (int c = 0; c < NCLASSES; ++c) red[c][tid] += red[c][tid + s];
    }
    __syncthreads();
  }
  if (tid < NCLASSES) out[tid] = red[tid][0];
}

// ---- launch --------------------------------------------------------------
extern "C" void kernel_launch(void* const* d_in, const int* in_sizes, int n_in,
                              void* d_out, int out_size, void* d_ws, size_t ws_size,
                              hipStream_t stream) {
  const float* Vp   = (const float*)d_in[0]; // [N, D]
  const float* Ut   = (const float*)d_in[1]; // [D', D]
  const float* Wmta = (const float*)d_in[2]; // [1, D']
  const float* W    = (const float*)d_in[3]; // [C, D]

  float* z     = (float*)d_ws;               // [N]
  float* a     = z + N_TOTAL;                // [N]
  float* tpart = a + N_TOTAL;                // [NCHUNKS, D]
  unsigned short* Abuf = (unsigned short*)(tpart + (size_t)NCHUNKS * DIM);

  const size_t need_base = (size_t)(2 * N_TOTAL + NCHUNKS * DIM) * sizeof(float);
  const size_t need_full = need_base + (size_t)DPRIME * DIM * sizeof(unsigned short);

  if (ws_size >= need_full) {
    // Pre-pack U_t -> bf16 fragment order: halves L2 traffic in the hot loop
    // and removes all A-side f32->bf16 conversions from K1.
    k_pack_ut<<<(DPRIME/16) * (DIM/32) * 32 / 256, 256, 0, stream>>>(Ut, Abuf);
    k_gemm_tanh_z<true><<<N_TOTAL / 64, 256, 0, stream>>>(Vp, Ut, Abuf, Wmta, z);
  } else {
    k_gemm_tanh_z<false><<<N_TOTAL / 64, 256, 0, stream>>>(Vp, Ut, Abuf, Wmta, z);
  }
  k_softmax<<<1, 1024, 0, stream>>>(z, a);
  k_pool<<<dim3(DIM / 256, NCHUNKS), 256, 0, stream>>>(Vp, a, tpart);
  k_final<<<1, 256, 0, stream>>>(tpart, W, (float*)d_out);
}
